// LBilinear_88622355186230
// MI455X (gfx1250) — matrix operand order repaired and checked by reading, not compiled, not run on hardware
//
#include <hip/hip_runtime.h>
#include <math.h>

// LBilinear: out[b,o,d,h,w] = atan2(Im,Re) + bias[o]
//   Re = c^T W[o] c - s^T W[o] s ; Im = c^T W[o] s + s^T W[o] c ; c=cos(x), s=sin(x)
// v2: A (weights, f16) fully register-resident per wave (8 o's/wave, 16 v16h frags),
//     cos/sin shared per block in LDS, 32 back-to-back WMMAs per tile per wave,
//     vector (v8f) epilogue to avoid register-pairing moves.

#define B_    16
#define CIN   32
#define COUT  32
#define PPB   8192                 // DIRS*H*W pixels per batch element
#define NTILE 8192                 // 16-pixel tiles total (131072/16)
#define NW    4                    // waves per block (wave32)
#define TPB   (NW * 32)
#define TILES_PER_BLOCK 4
#define GRID  (NTILE / TILES_PER_BLOCK)

typedef _Float16 v16h __attribute__((ext_vector_type(16)));
typedef _Float16 v8h  __attribute__((ext_vector_type(8)));
typedef float    v8f  __attribute__((ext_vector_type(8)));

__global__ __launch_bounds__(TPB) void lbilinear_wmma2(
    const float* __restrict__ x, const float* __restrict__ weight,
    const float* __restrict__ bias, float* __restrict__ out)
{
  // cos/sin in B-fragment layout: [cos|sin][p][k=c], k contiguous (64B per column)
  __shared__ __attribute__((aligned(32))) _Float16 csH[2][16][32];

  const int tid  = threadIdx.x;
  const int wave = tid >> 5;
  const int lane = tid & 31;
  const int p    = lane & 15;   // pixel within tile / WMMA N index
  const int hi   = lane >> 4;   // half-wave: selects K/M sub-ranges
  const int ow   = wave * 8;    // this wave's first output channel

  // ---- register-resident A fragments (loaded once, f32 -> f16) ----
  // Wstack row m = (o*2+par)*16 + p  maps to weight flat offset m*32.
  // Lane layout (16-bit A 16x32): elems 0-15 = K 16*hi..+15 (contiguous, matches B).
  v16h A[8][2];
#pragma unroll
  for (int j = 0; j < 8; ++j) {
#pragma unroll
    for (int par = 0; par < 2; ++par) {
      const int m = ((ow + j) * 2 + par) * 16 + p;
      const float4* rp = (const float4*)(weight + (size_t)m * CIN);
      float4 c0 = rp[4 * hi];
      float4 c1 = rp[4 * hi + 1];
      float4 c2 = rp[4 * hi + 2];
      float4 c3 = rp[4 * hi + 3];
      v16h a;
      a[0]  = (_Float16)c0.x; a[1]  = (_Float16)c0.y; a[2]  = (_Float16)c0.z; a[3]  = (_Float16)c0.w;
      a[4]  = (_Float16)c1.x; a[5]  = (_Float16)c1.y; a[6]  = (_Float16)c1.z; a[7]  = (_Float16)c1.w;
      a[8]  = (_Float16)c2.x; a[9]  = (_Float16)c2.y; a[10] = (_Float16)c2.z; a[11] = (_Float16)c2.w;
      a[12] = (_Float16)c3.x; a[13] = (_Float16)c3.y; a[14] = (_Float16)c3.z; a[15] = (_Float16)c3.w;
      A[j][par] = a;
    }
  }
  const float biasv = bias[ow];  // touch bias early (uniform, cached); per-o reload below

  for (int t = 0; t < TILES_PER_BLOCK; ++t) {
    const int    tileId = blockIdx.x * TILES_PER_BLOCK + t;
    const int    b      = tileId >> 9;            // 512 tiles per batch element
    const int    q      = (tileId & 511) << 4;
    const size_t xbase  = (size_t)b * CIN * PPB + q;

    // warm next pixels of each channel row (global_prefetch_b8)
    __builtin_prefetch(&x[xbase + 16 + (size_t)lane * PPB], 0, 1);

    // ---- cooperative cos/sin: 32 channels x 16 pixels, 4 values per thread ----
    {
      const int pt = tid & 15;
      const int cb = (tid >> 4) * 4;
#pragma unroll
      for (int i = 0; i < 4; ++i) {
        const int c  = cb + i;
        float xv = x[xbase + (size_t)c * PPB + pt];   // 64B-coalesced per 16 lanes
        float sn, cs;
        __sincosf(xv, &sn, &cs);
        csH[0][pt][c] = (_Float16)cs;
        csH[1][pt][c] = (_Float16)sn;
      }
    }
    __syncthreads();

    // ---- B fragments (K contiguous per lane) + epilogue coefficients ----
    v16h Bc = *(const v16h*)&csH[0][p][hi * 16];
    v16h Bs = *(const v16h*)&csH[1][p][hi * 16];
    v8h ch0 = *(const v8h*)&csH[0][p][8 * hi];        // c = r + 8*hi        (par 0)
    v8h ch1 = *(const v8h*)&csH[0][p][16 + 8 * hi];   // c = 16 + r + 8*hi   (par 1)
    v8h sh0 = *(const v8h*)&csH[1][p][8 * hi];
    v8h sh1 = *(const v8h*)&csH[1][p][16 + 8 * hi];
    v8f cv0, cv1, sv0, sv1;
#pragma unroll
    for (int r = 0; r < 8; ++r) {
      cv0[r] = (float)ch0[r]; cv1[r] = (float)ch1[r];
      sv0[r] = (float)sh0[r]; sv1[r] = (float)sh1[r];
    }

    const size_t obase = (size_t)b * COUT * PPB + q;

    // ---- 8 outputs per wave, 4 WMMAs each, all A operands in registers ----
#pragma unroll
    for (int j = 0; j < 8; ++j) {
      v8f U0 = {}, V0 = {}, U1 = {}, V1 = {};
      U0 = __builtin_amdgcn_wmma_f32_16x16x32_f16(false, A[j][0], false, Bc, (short)0, U0, false, false);
      V0 = __builtin_amdgcn_wmma_f32_16x16x32_f16(false, A[j][0], false, Bs, (short)0, V0, false, false);
      U1 = __builtin_amdgcn_wmma_f32_16x16x32_f16(false, A[j][1], false, Bc, (short)0, U1, false, false);
      V1 = __builtin_amdgcn_wmma_f32_16x16x32_f16(false, A[j][1], false, Bs, (short)0, V1, false, false);

      // diagonal extraction, elementwise over the 8 accumulator lanes-rows
      v8f reV = cv0 * U0 + cv1 * U1 - (sv0 * V0 + sv1 * V1);
      v8f imV = cv0 * V0 + cv1 * V1 + (sv0 * U0 + sv1 * U1);

      float re = ((reV[0] + reV[1]) + (reV[2] + reV[3])) + ((reV[4] + reV[5]) + (reV[6] + reV[7]));
      float im = ((imV[0] + imV[1]) + (imV[2] + imV[3])) + ((imV[4] + imV[5]) + (imV[6] + imV[7]));
      re += __shfl_xor(re, 16, 32);  // fold the two half-wave c-ranges (same pixel p)
      im += __shfl_xor(im, 16, 32);

      const int o = ow + j;
      float ang = atan2f(im, re) + bias[o];
      if (lane < 16) out[obase + (size_t)o * PPB + p] = ang;
    }
    __syncthreads();  // protect csH before next tile overwrites it
  }
  (void)biasv;
}

extern "C" void kernel_launch(void* const* d_in, const int* in_sizes, int n_in,
                              void* d_out, int out_size, void* d_ws, size_t ws_size,
                              hipStream_t stream) {
  const float* x    = (const float*)d_in[0];
  const float* w    = (const float*)d_in[1];
  const float* bias = (const float*)d_in[2];
  float*       out  = (float*)d_out;

  dim3 block(TPB);   // 4 wave32 waves; wave w owns output channels [8w, 8w+8)
  dim3 grid(GRID);   // 2048 blocks x 4 tiles each = 8192 sixteen-pixel tiles
  hipLaunchKernelGGL(lbilinear_wmma2, grid, block, 0, stream, x, w, bias, out);
}
